// InteractNet_14594298871826
// MI455X (gfx1250) — matrix operand order repaired and checked by reading, not compiled, not run on hardware
//
#include <hip/hip_runtime.h>
#include <math.h>

// ---------------------------------------------------------------------------
// Frequency-stage network for gfx1250 (MI455X).
// Heavy part (NRN MLP, ~74 GFLOP) runs on v_wmma_f32_16x16x32_f16 with
// spill-free register tiling: 4 independent accumulators per wave, A-fragments
// as two aligned ds_load_b128, B-fragments pre-packed for contiguous b128.
// ---------------------------------------------------------------------------

typedef __attribute__((ext_vector_type(16))) _Float16 v16h;
typedef __attribute__((ext_vector_type(8)))  float    v8f;

union F16x16 {
    uint4    q[2];
    _Float16 h[16];
    v16h     v;
};

#define Hh 256
#define Wd 256
#define Wf 129
#define Q  33024            // Hh*Wf
#define NC 64
#define ASTRIDE 584         // layer-0 input row stride (halfs), 16B aligned
#define CSTRIDE 264         // activation row stride (halfs): 132 dwords = 4 mod 64 banks

__device__ __forceinline__ float lrelu(float x) { return x > 0.f ? x : 0.1f * x; }

// ------------------------- 256-pt radix-2 FFT in LDS -----------------------
__device__ __forceinline__ void fft256(float* sre, float* sim, int tid, float sign) {
#pragma unroll
    for (int s = 1; s <= 8; ++s) {
        int m = 1 << s, half = m >> 1;
        int j  = tid & (half - 1);
        int i1 = ((tid >> (s - 1)) << s) + j;
        int i2 = i1 + half;
        float ang = sign * 6.283185307179586f * (float)j / (float)m;
        float wr = cosf(ang), wi = sinf(ang);
        __syncthreads();
        float br = sre[i2], bi = sim[i2];
        float tr = br * wr - bi * wi;
        float ti = br * wi + bi * wr;
        float ar = sre[i1], ai = sim[i1];
        sre[i1] = ar + tr; sim[i1] = ai + ti;
        sre[i2] = ar - tr; sim[i2] = ai - ti;
    }
    __syncthreads();
}

__global__ void k_fft_row(const float* __restrict__ x, float* __restrict__ Xre,
                          float* __restrict__ Xim) {
    __shared__ float sre[256], sim[256];
    size_t row = blockIdx.x;
    const float* xr = x + row * Wd;
    int tid = threadIdx.x;
    for (int i = tid; i < 256; i += 128) {
        int r = (int)(__brev((unsigned)i) >> 24);
        sre[r] = xr[i]; sim[r] = 0.f;
    }
    fft256(sre, sim, tid, -1.f);
    for (int k = tid; k <= 128; k += 128) {
        Xre[row * Wf + k] = sre[k];
        Xim[row * Wf + k] = sim[k];
    }
}

__global__ void k_fft_col(float* __restrict__ Re, float* __restrict__ Im,
                          float sign, float scale) {
    __shared__ float sre[256], sim[256];
    int n = blockIdx.x;
    int bc = n / Wf, xcol = n - bc * Wf;
    size_t base = (size_t)bc * Q + xcol;
    int tid = threadIdx.x;
    for (int i = tid; i < 256; i += 128) {
        int r = (int)(__brev((unsigned)i) >> 24);
        sre[r] = Re[base + (size_t)i * Wf];
        sim[r] = Im[base + (size_t)i * Wf];
    }
    fft256(sre, sim, tid, sign);
    for (int i = tid; i < 256; i += 128) {
        Re[base + (size_t)i * Wf] = sre[i] * scale;
        Im[base + (size_t)i * Wf] = sim[i] * scale;
    }
}

__global__ void k_irfft_row(const float* __restrict__ Sre, const float* __restrict__ Sim,
                            float* __restrict__ out) {
    __shared__ float sre[256], sim[256];
    size_t row = blockIdx.x;
    int tid = threadIdx.x;
    for (int i = tid; i < 256; i += 128) {
        int r = (int)(__brev((unsigned)i) >> 24);
        float re, im;
        if (i <= 128) { re = Sre[row * Wf + i];         im =  Sim[row * Wf + i]; }
        else          { re = Sre[row * Wf + (256 - i)]; im = -Sim[row * Wf + (256 - i)]; }
        sre[r] = re; sim[r] = im;
    }
    fft256(sre, sim, tid, 1.f);
    for (int i = tid; i < 256; i += 128)
        out[row * Wd + i] = sre[i] * (1.f / 256.f);
}

// ---- phase/amp + two 64->64 1x1-conv stacks -------------------------------
__global__ void k_phase_amp(const float* __restrict__ Xre, const float* __restrict__ Xim,
                            const float* pw1, const float* pb1, const float* pw2, const float* pb2,
                            const float* aw1, const float* ab1, const float* aw2, const float* ab2,
                            float* __restrict__ PHA, float* __restrict__ AMP) {
    extern __shared__ float sm[];
    float* vin  = sm;
    float* vhid = sm + 128 * 65;
    int tid = threadIdx.x;
    int p = blockIdx.x * 128 + tid;
    if (p >= 2 * Q) return;
    int b = p / Q, pos = p - b * Q;
    size_t base = (size_t)b * NC * Q + pos;
    for (int br = 0; br < 2; ++br) {
        for (int c = 0; c < NC; ++c) {
            float re = Xre[base + (size_t)c * Q];
            float im = Xim[base + (size_t)c * Q];
            vin[tid * 65 + c] = br ? sqrtf(re * re + im * im) : atan2f(im, re);
        }
        const float* w1 = br ? aw1 : pw1; const float* b1 = br ? ab1 : pb1;
        const float* w2 = br ? aw2 : pw2; const float* b2 = br ? ab2 : pb2;
        for (int m = 0; m < NC; ++m) {
            float s = b1[m];
            for (int c = 0; c < NC; ++c) s += w1[m * NC + c] * vin[tid * 65 + c];
            vhid[tid * 65 + m] = lrelu(s);
        }
        float* OUT = br ? AMP : PHA;
        for (int o = 0; o < NC; ++o) {
            float s = b2[o];
            for (int m = 0; m < NC; ++m) s += w2[o * NC + m] * vhid[tid * 65 + m];
            OUT[base + (size_t)o * Q] = s;
        }
    }
}

// ---- weight pre-pack into per-lane WMMA B-fragment order (f16) + b0_eff ---
// B as packed here (KxN, contiguous k, 32x16 tile): lanes 0-15 -> N, K=0..15 ; lanes 16-31 -> K=16..31
// pack index u = ((nt*KS + ks)*32 + lane)*16 + e ;  K = ks*32 + (lane>>4)*16 + e
__device__ __forceinline__ void pack_one(int u, const float* W, _Float16* P,
                                         int KS, int Kstride, int Nout) {
    int e = u & 15;
    int lane = (u >> 4) & 31;
    int ks = (u >> 9) % KS;
    int nt = (u >> 9) / KS;
    int N = nt * 16 + (lane & 15);
    int K = ks * 32 + (lane >> 4) * 16 + e;
    float v = (N < Nout) ? W[(size_t)N * Kstride + K] : 0.f;
    P[u] = (_Float16)v;
}

__global__ void k_prep(const float* w0, const float* b0, const float* w1,
                       const float* w2, const float* w3,
                       _Float16* W0p, _Float16* W1p, _Float16* W2p, _Float16* W3p,
                       float* b0eff) {
    int t = blockIdx.x * 256 + threadIdx.x;
    if (t < 147456)            pack_one(t,          w0, W0p, 18, 612, 256);
    else if (t < 212992)       pack_one(t - 147456, w1, W1p,  8, 256, 256);
    else if (t < 278528)       pack_one(t - 212992, w2, W2p,  8, 256, 256);
    else if (t < 282624)       pack_one(t - 278528, w3, W3p,  8, 256, 3);
    else if (t < 282880) {
        int o = t - 282624;   // fold rel_cell==2.0 columns (610,611) into bias
        b0eff[o] = b0[o] + 2.f * (w0[o * 612 + 610] + w0[o * 612 + 611]);
    }
}

// ---- A-fragment: two aligned 16B LDS chunks per lane ----------------------
// A order used here (contiguous k, same as B pack): lane half hf gets K = base+hf*16+{0..15}
__device__ __forceinline__ v16h load_afrag(const _Float16* IN, int stride,
                                           int row, int ks, int hf) {
    const uint4* ap = (const uint4*)(IN + (size_t)row * stride + ks * 32 + hf * 16);
    F16x16 a;
    a.q[0] = ap[0];
    a.q[1] = ap[1];
    return a.v;
}

__device__ __forceinline__ v16h load_bfrag(const _Float16* Wb) {
    const uint4* wq = (const uint4*)Wb;
    F16x16 b;
    b.q[0] = wq[0];
    b.q[1] = wq[1];
    return b.v;
}

// ---- generic WMMA GEMM layer, N=256, 4 accumulators per pass --------------
// IN: LDS f16 [16 rows x STRIDE], OUT: LDS f16 [rows x CSTRIDE].
// 4 independent D->C chains keep the XDL pipe fed; ~90 VGPRs, no spills.
template <int KS, int STRIDE>
__device__ __forceinline__ void gemm_layer(const _Float16* __restrict__ IN,
                                           _Float16* __restrict__ OUT,
                                           const _Float16* __restrict__ Wp,
                                           const float* __restrict__ bias,
                                           int rbase, int ml, int hf, int lane) {
    const int row = rbase + ml;
    for (int ntg = 0; ntg < 4; ++ntg) {
        v8f a0 = {0.f,0.f,0.f,0.f,0.f,0.f,0.f,0.f};
        v8f a1 = a0, a2 = a0, a3 = a0;
#pragma unroll
        for (int ks = 0; ks < KS; ++ks) {
            v16h af = load_afrag(IN, STRIDE, row, ks, hf);
            const _Float16* wb = Wp + ((size_t)((ntg * 4) * KS + ks) * 32 + lane) * 16;
            __builtin_prefetch(wb + 512, 0, 1);
            a0 = __builtin_amdgcn_wmma_f32_16x16x32_f16(false, af, false,
                     load_bfrag(wb),                     (short)0, a0, false, false);
            a1 = __builtin_amdgcn_wmma_f32_16x16x32_f16(false, af, false,
                     load_bfrag(wb + (size_t)KS * 512),  (short)0, a1, false, false);
            a2 = __builtin_amdgcn_wmma_f32_16x16x32_f16(false, af, false,
                     load_bfrag(wb + (size_t)KS * 1024), (short)0, a2, false, false);
            a3 = __builtin_amdgcn_wmma_f32_16x16x32_f16(false, af, false,
                     load_bfrag(wb + (size_t)KS * 1536), (short)0, a3, false, false);
        }
#pragma unroll
        for (int u = 0; u < 4; ++u) {
            v8f acc = (u == 0) ? a0 : (u == 1) ? a1 : (u == 2) ? a2 : a3;
            int ncol = (ntg * 4 + u) * 16 + ml;
            float bb = bias[ncol];
#pragma unroll
            for (int r = 0; r < 8; ++r) {
                float vv = acc[r] + bb;
                vv = vv > 0.f ? vv : 0.f;
                OUT[(size_t)(rbase + hf * 8 + r) * CSTRIDE + ncol] = (_Float16)vv;
            }
        }
    }
}

// ---- fused NRN MLP: unfold(576) -> 256 -> 256 -> 256 -> 3 -----------------
// grid = 2064 blocks x 128 threads (4 waves, 16 rows each, 64 rows/block)
// dyn LDS: A0 64x584 + ACT1/ACT2 64x264 f16 = 142336 B
__global__ void k_mlp(const float* __restrict__ PHA, const float* __restrict__ AMP,
                      const _Float16* __restrict__ W0p, const _Float16* __restrict__ W1p,
                      const _Float16* __restrict__ W2p, const _Float16* __restrict__ W3p,
                      const float* __restrict__ b0eff, const float* __restrict__ b1,
                      const float* __restrict__ b2, const float* __restrict__ b3,
                      float* __restrict__ NRNp, float* __restrict__ NRNa) {
    extern __shared__ char smem[];
    _Float16* A0   = (_Float16*)smem;            // 64 x ASTRIDE
    _Float16* ACT1 = A0 + 64 * ASTRIDE;          // 64 x CSTRIDE
    _Float16* ACT2 = ACT1 + 64 * CSTRIDE;        // 64 x CSTRIDE

    int tid = threadIdx.x;
    int base_row = blockIdx.x * 64;              // rows: ((branch*2+b)*Q + pos)
    int grp = base_row / Q;                      // uniform per block (64 | Q)
    int branch = grp >> 1, b = grp & 1;
    int pos0 = base_row - grp * Q;
    const float* Fb = (branch ? AMP : PHA) + (size_t)b * NC * Q;

    // stage 3x3 unfold (channel-major: u = c*9 + (i*3+j)) as f16
    for (int e = tid; e < 64 * 576; e += 128) {
        int pr = e / 576, u = e - pr * 576;
        int pos = pos0 + pr;
        int y = pos / Wf, xx = pos - y * Wf;
        int c = u / 9, pp = u - c * 9;
        int yy = y + pp / 3 - 1, xj = xx + pp % 3 - 1;
        float v = 0.f;
        if ((unsigned)yy < (unsigned)Hh && (unsigned)xj < (unsigned)Wf)
            v = Fb[((size_t)c * Hh + yy) * Wf + xj];
        A0[pr * ASTRIDE + u] = (_Float16)v;
    }
    __syncthreads();

    int wave = tid >> 5, lane = tid & 31;
    int ml = lane & 15, hf = lane >> 4;
    int rbase = wave * 16;

    gemm_layer<18, ASTRIDE>(A0,   ACT1, W0p, b0eff, rbase, ml, hf, lane);
    gemm_layer< 8, CSTRIDE>(ACT1, ACT2, W1p, b1,    rbase, ml, hf, lane);
    gemm_layer< 8, CSTRIDE>(ACT2, ACT1, W2p, b2,    rbase, ml, hf, lane);

    // ---- layer 3: K=256 -> 3 outputs (one padded n-tile, no relu) ----
    {
        v8f acc = {0.f,0.f,0.f,0.f,0.f,0.f,0.f,0.f};
#pragma unroll
        for (int ks = 0; ks < 8; ++ks) {
            v16h af = load_afrag(ACT1, CSTRIDE, rbase + ml, ks, hf);
            acc = __builtin_amdgcn_wmma_f32_16x16x32_f16(false, af, false,
                      load_bfrag(W3p + ((size_t)ks * 32 + lane) * 16),
                      (short)0, acc, false, false);
        }
        if (ml < 3) {
            float bb = b3[ml];
            float* OUT = branch ? NRNa : NRNp;
#pragma unroll
            for (int r = 0; r < 8; ++r) {
                int pos = pos0 + rbase + hf * 8 + r;
                OUT[(size_t)(b * 3 + ml) * Q + pos] = acc[r] + bb;
            }
        }
    }
}

// ---- 3->64->64 conv stacks on both NRN outputs, recombine to spectrum -----
__global__ void k_nconv(const float* __restrict__ NRNp, const float* __restrict__ NRNa,
                        const float* pnw1, const float* pnb1, const float* pnw2, const float* pnb2,
                        const float* anw1, const float* anb1, const float* anw2, const float* anb2,
                        float* __restrict__ T1re, float* __restrict__ T1im) {
    extern __shared__ float sm[];
    float* tmp  = sm;
    float* bufP = sm + 128 * 65;
    float* bufA = sm + 2 * 128 * 65;
    int tid = threadIdx.x;
    int p = blockIdx.x * 128 + tid;
    if (p >= 2 * Q) return;
    int b = p / Q, pos = p - b * Q;
    for (int br = 0; br < 2; ++br) {
        const float* IN = br ? NRNa : NRNp;
        const float* w1 = br ? anw1 : pnw1; const float* b1 = br ? anb1 : pnb1;
        const float* w2 = br ? anw2 : pnw2; const float* b2 = br ? anb2 : pnb2;
        float v0 = IN[(size_t)(b * 3 + 0) * Q + pos];
        float v1 = IN[(size_t)(b * 3 + 1) * Q + pos];
        float v2 = IN[(size_t)(b * 3 + 2) * Q + pos];
        for (int o = 0; o < NC; ++o)
            tmp[tid * 65 + o] = lrelu(b1[o] + w1[o * 3 + 0] * v0 + w1[o * 3 + 1] * v1 +
                                      w1[o * 3 + 2] * v2);
        float* BUF = br ? bufA : bufP;
        for (int o = 0; o < NC; ++o) {
            float s = b2[o];
            for (int m = 0; m < NC; ++m) s += w2[o * NC + m] * tmp[tid * 65 + m];
            BUF[tid * 65 + o] = s;
        }
    }
    for (int o = 0; o < NC; ++o) {
        float ph = bufP[tid * 65 + o], am = bufA[tid * 65 + o];
        T1re[(size_t)(b * NC + o) * Q + pos] = am * cosf(ph);
        T1im[(size_t)(b * NC + o) * Q + pos] = am * sinf(ph);
    }
}

// ---- complex GAP over spectrum --------------------------------------------
__global__ void k_reduce_yg(const float* __restrict__ Xre, const float* __restrict__ Xim,
                            float* __restrict__ ygre, float* __restrict__ ygim) {
    __shared__ float sr[256], si[256];
    int bc = blockIdx.x;
    size_t base = (size_t)bc * Q;
    float ar = 0.f, ai = 0.f;
    for (int i = threadIdx.x; i < Q; i += 256) { ar += Xre[base + i]; ai += Xim[base + i]; }
    int t = threadIdx.x;
    sr[t] = ar; si[t] = ai;
    __syncthreads();
    for (int st = 128; st > 0; st >>= 1) {
        if (t < st) { sr[t] += sr[t + st]; si[t] += si[t + st]; }
        __syncthreads();
    }
    if (t == 0) { ygre[bc] = sr[0] / (float)Q; ygim[bc] = si[0] / (float)Q; }
}

// ---- channel-branch complex value per (b,c) -------------------------------
__global__ void k_yvec(const float* __restrict__ ygre, const float* __restrict__ ygim,
                       const float* cpw, const float* cpb, const float* caw, const float* cab,
                       float* __restrict__ yvre, float* __restrict__ yvim) {
    int t = threadIdx.x;
    if (t >= 128) return;
    int b = t / NC, o = t - b * NC;
    float sp = cpb[o], sa = cab[o];
    for (int c = 0; c < NC; ++c) {
        float r = ygre[b * NC + c];
        sp += cpw[o * NC + c] * r;
        sa += caw[o * NC + c] * r;
    }
    float re = ygre[t], im = ygim[t];
    float yp = atan2f(im, re) * lrelu(sp);
    float ya = sqrtf(re * re + im * im) * lrelu(sa);
    yvre[t] = ya * cosf(yp);
    yvim[t] = ya * sinf(yp);
}

// ---- irfft2 of constant spectrum: only y==0 row nonzero -------------------
__global__ void k_ychrow(const float* __restrict__ yvre, const float* __restrict__ yvim,
                         float* __restrict__ ychrow) {
    int bc = blockIdx.x;
    float vr = yvre[bc], vi = yvim[bc];
    int x = threadIdx.x;
    float s = 0.f;
    for (int k = 0; k < 256; ++k) {
        float im = (k <= 128) ? vi : -vi;   // hermitian completion
        float ang = 6.283185307179586f * (float)((k * x) & 255) / 256.f;
        s += vr * cosf(ang) - im * sinf(ang);
    }
    ychrow[bc * 256 + x] = s / 256.f;
}

// ---- fold y_ch row through fin1 second half -------------------------------
__global__ void k_t2(const float* __restrict__ ychrow, const float* __restrict__ f1w,
                     float* __restrict__ t2) {
    int t = blockIdx.x * 256 + threadIdx.x;       // t = (b*64+o)*256 + x
    int x = t & 255, o = (t >> 8) & 63, b = t >> 14;
    float s = 0.f;
    for (int c = 0; c < NC; ++c)
        s += f1w[o * 128 + NC + c] * ychrow[(b * NC + c) * 256 + x];
    t2[t] = s;
}

// ---- final: spa = fin1([x_spatial, y_ch]) + x ; out = fin2(spa) -----------
__global__ void k_final(const float* __restrict__ xsp, const float* __restrict__ t2,
                        const float* __restrict__ xin,
                        const float* f1w, const float* f1b,
                        const float* f2w, const float* f2b,
                        float* __restrict__ out) {
    extern __shared__ float sm[];
    float* xs  = sm;
    float* spa = sm + 128 * 65;
    int tid = threadIdx.x;
    int p = blockIdx.x * 128 + tid;
    if (p >= 2 * Hh * Wd) return;
    int b = p >> 16, rem = p & 65535;
    int y = rem >> 8, xx = rem & 255;
    size_t pix = (size_t)y * Wd + xx;
    for (int c = 0; c < NC; ++c)
        xs[tid * 65 + c] = xsp[(size_t)(b * NC + c) * (Hh * Wd) + pix];
    for (int o = 0; o < NC; ++o) {
        float s = f1b[o];
        for (int c = 0; c < NC; ++c) s += f1w[o * 128 + c] * xs[tid * 65 + c];
        if (y == 0) s += t2[(b * NC + o) * 256 + xx];
        s += xin[(size_t)(b * NC + o) * (Hh * Wd) + pix];
        spa[tid * 65 + o] = s;
    }
    for (int ch = 0; ch < 3; ++ch) {
        float s = f2b[ch];
        for (int o = 0; o < NC; ++o) s += f2w[ch * NC + o] * spa[tid * 65 + o];
        out[(size_t)(b * 3 + ch) * (Hh * Wd) + pix] = s;
    }
}

// ---------------------------------------------------------------------------
extern "C" void kernel_launch(void* const* d_in, const int* in_sizes, int n_in,
                              void* d_out, int out_size, void* d_ws, size_t ws_size,
                              hipStream_t stream) {
    (void)in_sizes; (void)n_in; (void)out_size; (void)ws_size;
    const float* x    = (const float*)d_in[0];
    const float* pw1  = (const float*)d_in[1];  const float* pb1 = (const float*)d_in[2];
    const float* pw2  = (const float*)d_in[3];  const float* pb2 = (const float*)d_in[4];
    const float* aw1  = (const float*)d_in[5];  const float* ab1 = (const float*)d_in[6];
    const float* aw2  = (const float*)d_in[7];  const float* ab2 = (const float*)d_in[8];
    const float* w0   = (const float*)d_in[9];  const float* b0  = (const float*)d_in[10];
    const float* w1   = (const float*)d_in[11]; const float* b1  = (const float*)d_in[12];
    const float* w2   = (const float*)d_in[13]; const float* b2  = (const float*)d_in[14];
    const float* w3   = (const float*)d_in[15]; const float* b3  = (const float*)d_in[16];
    const float* pnw1 = (const float*)d_in[17]; const float* pnb1 = (const float*)d_in[18];
    const float* pnw2 = (const float*)d_in[19]; const float* pnb2 = (const float*)d_in[20];
    const float* anw1 = (const float*)d_in[21]; const float* anb1 = (const float*)d_in[22];
    const float* anw2 = (const float*)d_in[23]; const float* anb2 = (const float*)d_in[24];
    const float* cpw  = (const float*)d_in[25]; const float* cpb = (const float*)d_in[26];
    const float* caw  = (const float*)d_in[27]; const float* cab = (const float*)d_in[28];
    const float* f1w  = (const float*)d_in[29]; const float* f1b = (const float*)d_in[30];
    const float* f2w  = (const float*)d_in[31]; const float* f2b = (const float*)d_in[32];
    float* out = (float*)d_out;

    // ---- workspace carve (float units; all chunks are 32B-aligned) ----
    float* W = (float*)d_ws;
    const size_t SPEC = (size_t)2 * NC * Hh * Wf;   // 4,227,072
    size_t off = 0;
    float* Xre  = W + off; off += SPEC;
    float* Xim  = W + off; off += SPEC;
    float* PHA  = W + off; off += SPEC;
    float* AMPb = W + off; off += SPEC;
    float* T1re = W + off; off += SPEC;
    float* T1im = W + off; off += SPEC;
    float* XSP  = W + off; off += (size_t)2 * NC * Hh * Wd;  // 8,388,608
    float* NRNp = W + off; off += (size_t)2 * 3 * Q;
    float* NRNa = W + off; off += (size_t)2 * 3 * Q;
    float* ygre = W + off; off += 128;
    float* ygim = W + off; off += 128;
    float* yvre = W + off; off += 128;
    float* yvim = W + off; off += 128;
    float* ychr = W + off; off += (size_t)2 * NC * 256;
    float* t2b  = W + off; off += (size_t)2 * NC * 256;
    float* b0ef = W + off; off += 256;
    _Float16* W0p = (_Float16*)(W + off);
    _Float16* W1p = W0p + 147456;
    _Float16* W2p = W1p + 65536;
    _Float16* W3p = W2p + 65536;

    // 0) weight packing + folded bias
    k_prep<<<1105, 256, 0, stream>>>(w0, b0, w1, w2, w3, W0p, W1p, W2p, W3p, b0ef);
    // 1) rfft2 = row rfft then column fft
    k_fft_row<<<2 * NC * Hh, 128, 0, stream>>>(x, Xre, Xim);
    k_fft_col<<<2 * NC * Wf, 128, 0, stream>>>(Xre, Xim, -1.f, 1.f);
    // 2) phase/amp + conv stacks
    k_phase_amp<<<(2 * Q) / 128, 128, 2 * 128 * 65 * sizeof(float), stream>>>(
        Xre, Xim, pw1, pb1, pw2, pb2, aw1, ab1, aw2, ab2, PHA, AMPb);
    // 3) fused WMMA MLP (both branches, both batches)
    k_mlp<<<2064, 128, (64 * ASTRIDE + 2 * 64 * CSTRIDE) * sizeof(_Float16), stream>>>(
        PHA, AMPb, W0p, W1p, W2p, W3p, b0ef, b1, b2, b3, NRNp, NRNa);
    // 4) 3->64->64 stacks, recombine into spectrum
    k_nconv<<<(2 * Q) / 128, 128, 3 * 128 * 65 * sizeof(float), stream>>>(
        NRNp, NRNa, pnw1, pnb1, pnw2, pnb2, anw1, anb1, anw2, anb2, T1re, T1im);
    // 5) irfft2 = column ifft (1/256) then row irfft (1/256)
    k_fft_col<<<2 * NC * Wf, 128, 0, stream>>>(T1re, T1im, 1.f, 1.f / 256.f);
    k_irfft_row<<<2 * NC * Hh, 128, 0, stream>>>(T1re, T1im, XSP);
    // 6) channel branch (GAP -> complex value -> single nonzero row, folded)
    k_reduce_yg<<<2 * NC, 256, 0, stream>>>(Xre, Xim, ygre, ygim);
    k_yvec<<<1, 128, 0, stream>>>(ygre, ygim, cpw, cpb, caw, cab, yvre, yvim);
    k_ychrow<<<2 * NC, 256, 0, stream>>>(yvre, yvim, ychr);
    k_t2<<<128, 256, 0, stream>>>(ychr, f1w, t2b);
    // 7) final convs + residual
    k_final<<<(2 * Hh * Wd) / 128, 128, 2 * 128 * 65 * sizeof(float), stream>>>(
        XSP, t2b, x, f1w, f1b, f2w, f2b, out);
}
